// ViT_6940667150572
// MI455X (gfx1250) — matrix operand order repaired and checked by reading, not compiled, not run on hardware
//
#include <hip/hip_runtime.h>
#include <math.h>

// ---------------- ViT config ----------------
#define B_    64
#define C_    3
#define HI_   224
#define NP_   14
#define PS_   16
#define L_    8
#define D_    384
#define NH_   6
#define DH_   64
#define MLP_  1536
#define OUT_  1000
#define OUTP_ 1008          // padded classifier cols
#define S_    197
#define SP_   224           // padded key/token dim (7*32)
#define RP_   208           // padded attn row dim (13*16)
#define IND_  768
#define NTOK_   (B_ * S_)          // 12608 = 788*16
#define NPROWS_ (B_ * NP_ * NP_)   // 12544 = 784*16
#define MT_TOK  (NTOK_ / 16)       // 788
#define QT_     ((S_ + 15) / 16)   // 13

typedef __attribute__((ext_vector_type(16))) __bf16 v16bf;
typedef __attribute__((ext_vector_type(8)))  __bf16 v8bf;
typedef __attribute__((ext_vector_type(8)))  float  v8f;

// ---------------- WMMA ----------------
__device__ __forceinline__ v8f wmma_bf16(v16bf a, v16bf b, v8f c) {
  return __builtin_amdgcn_wmma_f32_16x16x32_bf16(false, a, false, b, (short)0, c, false, false);
}

// A fragment 16(M)x32(K) from row-major bf16: two contiguous 16B runs per lane.
__device__ __forceinline__ v16bf load_a16(const __bf16* __restrict__ p, long lda,
                                          int row0, int k0) {
  const int lane = threadIdx.x & 31;
  const __bf16* r = p + (long)(row0 + (lane & 15)) * lda + (k0 + ((lane >> 4) << 3));
  const v8bf u0 = *(const v8bf*)(r);
  const v8bf u1 = *(const v8bf*)(r + 16);
  return __builtin_shufflevector(u0, u1, 0,1,2,3,4,5,6,7,8,9,10,11,12,13,14,15);
}

// A fragment from row-major f32 (x / attn-scores style sources).
__device__ __forceinline__ v16bf load_a32(const float* __restrict__ p, long lda,
                                          int row0, int k0) {
  const int lane = threadIdx.x & 31;
  const float* r = p + (long)(row0 + (lane & 15)) * lda + (k0 + ((lane >> 4) << 3));
  const float4 u0 = *(const float4*)(r);
  const float4 u1 = *(const float4*)(r + 4);
  const float4 u2 = *(const float4*)(r + 16);
  const float4 u3 = *(const float4*)(r + 20);
  v16bf a;
  a[0]=(__bf16)u0.x;  a[1]=(__bf16)u0.y;  a[2]=(__bf16)u0.z;  a[3]=(__bf16)u0.w;
  a[4]=(__bf16)u1.x;  a[5]=(__bf16)u1.y;  a[6]=(__bf16)u1.z;  a[7]=(__bf16)u1.w;
  a[8]=(__bf16)u2.x;  a[9]=(__bf16)u2.y;  a[10]=(__bf16)u2.z; a[11]=(__bf16)u2.w;
  a[12]=(__bf16)u3.x; a[13]=(__bf16)u3.y; a[14]=(__bf16)u3.z; a[15]=(__bf16)u3.w;
  return a;
}

// B fragment 32(K)x16(N) from K-contiguous bf16 [Npad x K] (transposed weights,
// K-matrix read along D, or V stored transposed): two contiguous 16B runs per lane.
__device__ __forceinline__ v16bf load_bT16(const __bf16* __restrict__ p, long ldk,
                                           int k0, int n0) {
  const int lane = threadIdx.x & 31;
  const __bf16* r = p + (long)(n0 + (lane & 15)) * ldk + (k0 + ((lane >> 4) << 3));
  const v8bf u0 = *(const v8bf*)(r);
  const v8bf u1 = *(const v8bf*)(r + 16);
  return __builtin_shufflevector(u0, u1, 0,1,2,3,4,5,6,7,8,9,10,11,12,13,14,15);
}

// ---------------- weight transpose to bf16: src[b][K][N] -> dst[b][Npad][K] ----------------
__global__ void k_tr_bf16(const float* __restrict__ src, __bf16* __restrict__ dst,
                          int K, int N, int Npad, long total) {
  const long i = (long)blockIdx.x * 256 + threadIdx.x;
  if (i >= total) return;
  const int  k = (int)(i % K);
  const long t = i / K;
  const int  n = (int)(t % Npad);
  const int  b = (int)(t / Npad);
  float v = 0.0f;
  if (n < N) v = src[((long)b * K + k) * N + n];
  dst[i] = (__bf16)v;
}

// ---------------- patch embed ----------------
__device__ __forceinline__ const float* patch_ptr(const float* __restrict__ img, int m, int e) {
  const int b = m / (NP_ * NP_), p = m % (NP_ * NP_);
  const int py = p / NP_, px = p % NP_;
  const int c = e >> 8, ph = (e >> 4) & 15, pw = e & 15;
  return img + (((long)b * C_ + c) * HI_ + (py * PS_ + ph)) * HI_ + (px * PS_ + pw);
}

__global__ void k_patch_embed(const float* __restrict__ images,
                              const __bf16* __restrict__ WmapT,
                              const float* __restrict__ bmap,
                              const float* __restrict__ pos,
                              float* __restrict__ x) {
  const int NG = D_ / 64;                       // 6 groups of 4 N-tiles
  const int g  = blockIdx.x % NG;
  const int mt = blockIdx.x / NG;
  const int lane = threadIdx.x & 31;
  const int m  = mt * 16 + (lane & 15);
  v8f acc[4] = {};
  for (int k0 = 0; k0 < IND_; k0 += 32) {
    const int e = k0 + ((lane >> 4) << 3);
    const float* r0 = patch_ptr(images, m, e);
    const float* r1 = patch_ptr(images, m, e + 16);
    const float4 u0 = *(const float4*)(r0);
    const float4 u1 = *(const float4*)(r0 + 4);
    const float4 u2 = *(const float4*)(r1);
    const float4 u3 = *(const float4*)(r1 + 4);
    v16bf a;
    a[0]=(__bf16)u0.x;  a[1]=(__bf16)u0.y;  a[2]=(__bf16)u0.z;  a[3]=(__bf16)u0.w;
    a[4]=(__bf16)u1.x;  a[5]=(__bf16)u1.y;  a[6]=(__bf16)u1.z;  a[7]=(__bf16)u1.w;
    a[8]=(__bf16)u2.x;  a[9]=(__bf16)u2.y;  a[10]=(__bf16)u2.z; a[11]=(__bf16)u2.w;
    a[12]=(__bf16)u3.x; a[13]=(__bf16)u3.y; a[14]=(__bf16)u3.z; a[15]=(__bf16)u3.w;
#pragma unroll
    for (int t = 0; t < 4; ++t)
      acc[t] = wmma_bf16(a, load_bT16(WmapT, IND_, k0, g * 64 + t * 16), acc[t]);
  }
  const int mbase = mt * 16 + ((lane >> 4) << 3);
#pragma unroll
  for (int r = 0; r < 8; ++r) {
    const int mm = mbase + r;
    const int b = mm / (NP_ * NP_);
    const int s = (mm % (NP_ * NP_)) + 1;
#pragma unroll
    for (int t = 0; t < 4; ++t) {
      const int n = g * 64 + t * 16 + (lane & 15);
      x[((long)b * S_ + s) * D_ + n] = acc[t][r] + bmap[n] + pos[(long)s * D_ + n];
    }
  }
}

__global__ void k_cls(const float* __restrict__ cls, const float* __restrict__ pos,
                      float* __restrict__ x) {
  const int b = blockIdx.x, d = threadIdx.x;
  if (d < D_) x[(long)b * S_ * D_ + d] = cls[d] + pos[d];
}

// ---------------- LayerNorm (f32 in, bf16 out) ----------------
__device__ __forceinline__ float block_sum(float v, float* red) {
  const int t = threadIdx.x;
  red[t] = v; __syncthreads();
  for (int o = blockDim.x >> 1; o > 0; o >>= 1) {
    if (t < o) red[t] += red[t + o];
    __syncthreads();
  }
  const float r = red[0]; __syncthreads();
  return r;
}

__global__ void k_layernorm(const float* __restrict__ x, const float* __restrict__ g,
                            const float* __restrict__ b, __bf16* __restrict__ h) {
  __shared__ float red[128];
  const long t = blockIdx.x;
  const float* xr = x + t * D_;
  float s = 0.0f;
  for (int i = threadIdx.x; i < D_; i += 128) s += xr[i];
  const float mu = block_sum(s, red) * (1.0f / D_);
  float vs = 0.0f;
  for (int i = threadIdx.x; i < D_; i += 128) { const float d = xr[i] - mu; vs += d * d; }
  const float var = block_sum(vs, red) * (1.0f / D_);
  const float rstd = rsqrtf(var + 1e-5f);
  __bf16* hr = h + t * D_;
  for (int i = threadIdx.x; i < D_; i += 128)
    hr[i] = (__bf16)(g[i] * (xr[i] - mu) * rstd + b[i]);
}

// ---------------- fused Q/K/V (bf16 out; V stored transposed per (b,head)) ----------------
__global__ void k_qkv(const __bf16* __restrict__ h,
                      const __bf16* __restrict__ WqT, const float* __restrict__ bq,
                      const __bf16* __restrict__ WkT, const float* __restrict__ bk,
                      const __bf16* __restrict__ WvT, const float* __restrict__ bv,
                      __bf16* __restrict__ q, __bf16* __restrict__ kk,
                      __bf16* __restrict__ vT) {
  const int head = blockIdx.x % NH_;
  const int mt   = blockIdx.x / NH_;
  const __bf16* wq = WqT + (long)head * DH_ * DH_;
  const __bf16* wk = WkT + (long)head * DH_ * DH_;
  const __bf16* wv = WvT + (long)head * DH_ * DH_;
  v8f aq[4] = {}, ak[4] = {}, av[4] = {};
  for (int k0 = 0; k0 < DH_; k0 += 32) {
    const v16bf a = load_a16(h, D_, mt * 16, head * DH_ + k0);
#pragma unroll
    for (int t = 0; t < 4; ++t) {
      aq[t] = wmma_bf16(a, load_bT16(wq, DH_, k0, t * 16), aq[t]);
      ak[t] = wmma_bf16(a, load_bT16(wk, DH_, k0, t * 16), ak[t]);
      av[t] = wmma_bf16(a, load_bT16(wv, DH_, k0, t * 16), av[t]);
    }
  }
  const int lane  = threadIdx.x & 31;
  const int mbase = mt * 16 + ((lane >> 4) << 3);
#pragma unroll
  for (int r = 0; r < 8; ++r) {
    const int m = mbase + r;
    const int b = m / S_;
    const int s = m - b * S_;
#pragma unroll
    for (int t = 0; t < 4; ++t) {
      const int n   = t * 16 + (lane & 15);
      const int col = head * DH_ + n;
      q[(long)m * D_ + col]  = (__bf16)(aq[t][r] + bq[head * DH_ + n]);
      kk[(long)m * D_ + col] = (__bf16)(ak[t][r] + bk[head * DH_ + n]);
      vT[((long)(b * NH_ + head) * DH_ + n) * SP_ + s] = (__bf16)(av[t][r] + bv[head * DH_ + n]);
    }
  }
}

// ---------------- scores = scale * Q @ K^T (f32 out for softmax) ----------------
__global__ void k_scores(const __bf16* __restrict__ q, const __bf16* __restrict__ kbuf,
                         float* __restrict__ scores) {
  int id = blockIdx.x;
  const int qt   = id % QT_; id /= QT_;
  const int head = id % NH_; id /= NH_;
  const int b    = id;
  const __bf16* qb = q    + (long)b * S_ * D_;
  const __bf16* kb = kbuf + (long)b * S_ * D_;
  v8f acc[QT_] = {};
  for (int e0 = 0; e0 < DH_; e0 += 32) {
    const v16bf a = load_a16(qb, D_, qt * 16, head * DH_ + e0);
#pragma unroll
    for (int kt = 0; kt < QT_; ++kt)
      acc[kt] = wmma_bf16(a, load_bT16(kb, D_, head * DH_ + e0, kt * 16), acc[kt]);
  }
  const int lane  = threadIdx.x & 31;
  const int mbase = qt * 16 + ((lane >> 4) << 3);
  float* srow = scores + ((long)(b * NH_ + head) * S_) * SP_;
#pragma unroll
  for (int kt = 0; kt < QT_; ++kt) {
    const int n = kt * 16 + (lane & 15);
#pragma unroll
    for (int r = 0; r < 8; ++r) {
      const int m = mbase + r;
      if (m < S_ && n < S_) srow[(long)m * SP_ + n] = acc[kt][r] * 0.125f;
    }
  }
}

// ---------------- softmax rows: f32 scores -> bf16 probs [RP_ x SP_] slices ----------------
__global__ void k_softmax_rows(const float* __restrict__ scores, __bf16* __restrict__ att) {
  __shared__ float red[128];
  const int bh = blockIdx.x / S_;
  const int m  = blockIdx.x - bh * S_;
  const float* p = scores + ((long)bh * S_ + m) * SP_;
  __bf16* o = att + ((long)bh * RP_ + m) * SP_;
  float mx = -3.0e38f;
  for (int i = threadIdx.x; i < S_; i += 128) mx = fmaxf(mx, p[i]);
  {
    const int t = threadIdx.x;
    red[t] = mx; __syncthreads();
    for (int of = 64; of > 0; of >>= 1) { if (t < of) red[t] = fmaxf(red[t], red[t + of]); __syncthreads(); }
    mx = red[0]; __syncthreads();
  }
  float sum = 0.0f;
  for (int i = threadIdx.x; i < S_; i += 128) sum += expf(p[i] - mx);
  const float tot = block_sum(sum, red);
  const float inv = 1.0f / tot;
  for (int i = threadIdx.x; i < S_; i += 128) o[i] = (__bf16)(expf(p[i] - mx) * inv);
  for (int i = S_ + threadIdx.x; i < SP_; i += 128) o[i] = (__bf16)0.0f;
}

// ---------------- x += attn @ V (all-bf16 fragments, unchecked K loop) ----------------
__global__ void k_av(const __bf16* __restrict__ att, const __bf16* __restrict__ vT,
                     float* __restrict__ x) {
  int id = blockIdx.x;
  const int qt   = id % QT_; id /= QT_;
  const int head = id % NH_; id /= NH_;
  const int b    = id;
  const __bf16* as = att + (long)(b * NH_ + head) * RP_ * SP_;
  const __bf16* vs = vT  + (long)(b * NH_ + head) * DH_ * SP_;
  v8f acc[4] = {};
  for (int k0 = 0; k0 < SP_; k0 += 32) {
    const v16bf a = load_a16(as, SP_, qt * 16, k0);
#pragma unroll
    for (int t = 0; t < 4; ++t)
      acc[t] = wmma_bf16(a, load_bT16(vs, SP_, k0, t * 16), acc[t]);
  }
  const int lane  = threadIdx.x & 31;
  const int mbase = qt * 16 + ((lane >> 4) << 3);
#pragma unroll
  for (int t = 0; t < 4; ++t) {
    const int n = head * DH_ + t * 16 + (lane & 15);
#pragma unroll
    for (int r = 0; r < 8; ++r) {
      const int m = mbase + r;
      if (m < S_) x[((long)b * S_ + m) * D_ + n] += acc[t][r];
    }
  }
}

// ---------------- MLP1 + exact GELU : 2 M-tiles x 8 N-tiles per wave ----------------
__global__ void k_mlp1(const __bf16* __restrict__ h, const __bf16* __restrict__ W1T,
                       const float* __restrict__ b1, __bf16* __restrict__ m1) {
  const int NG = MLP_ / 128;                    // 12
  const int g  = blockIdx.x % NG;
  const int m0 = (blockIdx.x / NG) * 32;
  v8f acc[2][8] = {};
  for (int k0 = 0; k0 < D_; k0 += 32) {
    const v16bf a0 = load_a16(h, D_, m0, k0);
    const v16bf a1 = load_a16(h, D_, m0 + 16, k0);
#pragma unroll
    for (int t = 0; t < 8; ++t) {
      const v16bf bfr = load_bT16(W1T, D_, k0, g * 128 + t * 16);
      acc[0][t] = wmma_bf16(a0, bfr, acc[0][t]);
      acc[1][t] = wmma_bf16(a1, bfr, acc[1][t]);
    }
  }
  const int lane = threadIdx.x & 31;
#pragma unroll
  for (int t = 0; t < 8; ++t) {
    const int n = g * 128 + t * 16 + (lane & 15);
    const float bias = b1[n];
#pragma unroll
    for (int u = 0; u < 2; ++u) {
      const int mbase = m0 + u * 16 + ((lane >> 4) << 3);
#pragma unroll
      for (int r = 0; r < 8; ++r) {
        const float val = acc[u][t][r] + bias;
        m1[(long)(mbase + r) * MLP_ + n] =
            (__bf16)(0.5f * val * (1.0f + erff(val * 0.70710678118654752f)));
      }
    }
  }
}

// ---------------- x += MLP2 : 2 M-tiles x 8 N-tiles per wave ----------------
__global__ void k_mlp2(const __bf16* __restrict__ m1, const __bf16* __restrict__ W2T,
                       const float* __restrict__ b2, float* __restrict__ x) {
  const int NG = D_ / 128;                      // 3
  const int g  = blockIdx.x % NG;
  const int m0 = (blockIdx.x / NG) * 32;
  v8f acc[2][8] = {};
  for (int k0 = 0; k0 < MLP_; k0 += 32) {
    const v16bf a0 = load_a16(m1, MLP_, m0, k0);
    const v16bf a1 = load_a16(m1, MLP_, m0 + 16, k0);
#pragma unroll
    for (int t = 0; t < 8; ++t) {
      const v16bf bfr = load_bT16(W2T, MLP_, k0, g * 128 + t * 16);
      acc[0][t] = wmma_bf16(a0, bfr, acc[0][t]);
      acc[1][t] = wmma_bf16(a1, bfr, acc[1][t]);
    }
  }
  const int lane = threadIdx.x & 31;
#pragma unroll
  for (int t = 0; t < 8; ++t) {
    const int n = g * 128 + t * 16 + (lane & 15);
    const float bias = b2[n];
#pragma unroll
    for (int u = 0; u < 2; ++u) {
      const int mbase = m0 + u * 16 + ((lane >> 4) << 3);
#pragma unroll
      for (int r = 0; r < 8; ++r)
        x[(long)(mbase + r) * D_ + n] += acc[u][t][r] + bias;
    }
  }
}

// ---------------- classifier head ----------------
__global__ void k_head(const float* __restrict__ x, const __bf16* __restrict__ WhT,
                       const float* __restrict__ bh, float* __restrict__ logits) {
  const int NT = OUTP_ / 16;                    // 63
  const int nt = blockIdx.x % NT;
  const int mt = blockIdx.x / NT;
  v8f acc = {};
  for (int k0 = 0; k0 < D_; k0 += 32) {
    const v16bf a = load_a32(x, (long)S_ * D_, mt * 16, k0);   // class-token rows
    acc = wmma_bf16(a, load_bT16(WhT, D_, k0, nt * 16), acc);
  }
  const int lane  = threadIdx.x & 31;
  const int n     = nt * 16 + (lane & 15);
  const int mbase = mt * 16 + ((lane >> 4) << 3);
#pragma unroll
  for (int r = 0; r < 8; ++r)
    if (n < OUT_) logits[(long)(mbase + r) * OUT_ + n] = acc[r] + bh[n];
}

__global__ void k_softmax_out(const float* __restrict__ logits, float* __restrict__ out) {
  __shared__ float red[256];
  const float* p = logits + (long)blockIdx.x * OUT_;
  float* o = out + (long)blockIdx.x * OUT_;
  float mx = -3.0e38f;
  for (int i = threadIdx.x; i < OUT_; i += 256) mx = fmaxf(mx, p[i]);
  {
    const int t = threadIdx.x;
    red[t] = mx; __syncthreads();
    for (int of = 128; of > 0; of >>= 1) { if (t < of) red[t] = fmaxf(red[t], red[t + of]); __syncthreads(); }
    mx = red[0]; __syncthreads();
  }
  float sum = 0.0f;
  for (int i = threadIdx.x; i < OUT_; i += 256) { const float e = expf(p[i] - mx); o[i] = e; sum += e; }
  red[threadIdx.x] = sum; __syncthreads();
  for (int of = 128; of > 0; of >>= 1) { if (threadIdx.x < of) red[threadIdx.x] += red[threadIdx.x + of]; __syncthreads(); }
  const float inv = 1.0f / red[0];
  for (int i = threadIdx.x; i < OUT_; i += 256) o[i] *= inv;
}

// ---------------- host launcher ----------------
extern "C" void kernel_launch(void* const* d_in, const int* in_sizes, int n_in,
                              void* d_out, int out_size, void* d_ws, size_t ws_size,
                              hipStream_t stream) {
  const float* images = (const float*)d_in[0];
  const float* Wmap   = (const float*)d_in[1];
  const float* bmap   = (const float*)d_in[2];
  const float* cls    = (const float*)d_in[3];
  const float* pos    = (const float*)d_in[4];
  const float* ln1s   = (const float*)d_in[5];
  const float* ln1b   = (const float*)d_in[6];
  const float* Wq     = (const float*)d_in[7];
  const float* bq     = (const float*)d_in[8];
  const float* Wk     = (const float*)d_in[9];
  const float* bk     = (const float*)d_in[10];
  const float* Wv     = (const float*)d_in[11];
  const float* bv     = (const float*)d_in[12];
  const float* ln2s   = (const float*)d_in[13];
  const float* ln2b   = (const float*)d_in[14];
  const float* W1     = (const float*)d_in[15];
  const float* b1     = (const float*)d_in[16];
  const float* W2     = (const float*)d_in[17];
  const float* b2     = (const float*)d_in[18];
  const float* Wh     = (const float*)d_in[19];
  const float* bh     = (const float*)d_in[20];

  // ---- f32 workspace ----
  const long XC  = (long)NTOK_ * D_;                     // 4,841,472
  const long SCN = (long)B_ * NH_ * S_ * SP_;            // f32 scores
  float* ws     = (float*)d_ws;
  float* x      = ws;
  float* scores = ws + XC;
  float* logits = scores + SCN;
  // ---- bf16 workspace (activations + transposed weights) ----
  __bf16* bws   = (__bf16*)(logits + (long)B_ * OUT_);
  __bf16* h16   = bws;                                   // NTOK*D
  __bf16* q16   = h16 + XC;                              // NTOK*D
  __bf16* k16   = q16 + XC;                              // NTOK*D (q edge-reads spill here)
  __bf16* vT16  = k16 + XC;                              // B*NH*DH*SP (k edge-reads spill here)
  __bf16* att16 = vT16 + (long)B_ * NH_ * DH_ * SP_;     // B*NH*RP*SP
  __bf16* m116  = att16 + (long)B_ * NH_ * RP_ * SP_;    // NTOK*MLP
  __bf16* WmapT = m116 + (long)NTOK_ * MLP_;             // 768*384
  __bf16* WqT   = WmapT + (long)IND_ * D_;
  __bf16* WkT   = WqT + (long)L_ * NH_ * DH_ * DH_;
  __bf16* WvT   = WkT + (long)L_ * NH_ * DH_ * DH_;
  __bf16* W1T   = WvT + (long)L_ * NH_ * DH_ * DH_;      // L*1536*384
  __bf16* W2T   = W1T + (long)L_ * MLP_ * D_;            // L*384*1536
  __bf16* WhT   = W2T + (long)L_ * D_ * MLP_;            // 1008*384

  const dim3 wv32(32);
  auto tr = [&](const float* s, __bf16* d, int K, int N, int Npad, int batch) {
    const long tot = (long)batch * Npad * K;
    k_tr_bf16<<<dim3((unsigned)((tot + 255) / 256)), dim3(256), 0, stream>>>(s, d, K, N, Npad, tot);
  };
  tr(Wmap, WmapT, IND_, D_, D_, 1);
  tr(Wq, WqT, DH_, DH_, DH_, L_ * NH_);
  tr(Wk, WkT, DH_, DH_, DH_, L_ * NH_);
  tr(Wv, WvT, DH_, DH_, DH_, L_ * NH_);
  tr(W1, W1T, D_, MLP_, MLP_, L_);
  tr(W2, W2T, MLP_, D_, D_, L_);
  tr(Wh, WhT, D_, OUT_, OUTP_, 1);

  // embed
  k_patch_embed<<<dim3((NPROWS_ / 16) * (D_ / 64)), wv32, 0, stream>>>(images, WmapT, bmap, pos, x);
  k_cls<<<dim3(B_), dim3(D_), 0, stream>>>(cls, pos, x);

  for (int l = 0; l < L_; ++l) {
    const long wqo = (long)l * NH_ * DH_ * DH_, bqo = (long)l * NH_ * DH_;
    k_layernorm<<<dim3(NTOK_), dim3(128), 0, stream>>>(x, ln1s + (long)l * D_, ln1b + (long)l * D_, h16);
    k_qkv<<<dim3(MT_TOK * NH_), wv32, 0, stream>>>(h16,
        WqT + wqo, bq + bqo, WkT + wqo, bk + bqo, WvT + wqo, bv + bqo, q16, k16, vT16);
    k_scores<<<dim3(B_ * NH_ * QT_), wv32, 0, stream>>>(q16, k16, scores);
    k_softmax_rows<<<dim3(B_ * NH_ * S_), dim3(128), 0, stream>>>(scores, att16);
    k_av<<<dim3(B_ * NH_ * QT_), wv32, 0, stream>>>(att16, vT16, x);
    k_layernorm<<<dim3(NTOK_), dim3(128), 0, stream>>>(x, ln2s + (long)l * D_, ln2b + (long)l * D_, h16);
    k_mlp1<<<dim3((MT_TOK / 2) * (MLP_ / 128)), wv32, 0, stream>>>(h16, W1T + (long)l * D_ * MLP_, b1 + (long)l * MLP_, m116);
    k_mlp2<<<dim3((MT_TOK / 2) * (D_ / 128)), wv32, 0, stream>>>(m116, W2T + (long)l * MLP_ * D_, b2 + (long)l * D_, x);
  }

  k_head<<<dim3((B_ / 16) * (OUTP_ / 16)), wv32, 0, stream>>>(x, WhT, bh, logits);
  k_softmax_out<<<dim3(B_), dim3(256), 0, stream>>>(logits, (float*)d_out);
}
